// StereoAttention_68977174774083
// MI455X (gfx1250) — matrix operand order repaired and checked by reading, not compiled, not run on hardware
//
#include <hip/hip_runtime.h>
#include <math.h>

// Problem constants (from reference)
#define NB      4
#define HH      120
#define WW      160
#define CC      128
#define TWO_C   256
#define KW      5            // conv width
#define MM      10           // attention channels
#define K_TOTAL (KW * TWO_C) // 1280 reduction length
#define KCH     (K_TOTAL/32) // 40 k-chunks of 32 for 16x16x32 WMMA
#define BLOCK   320          // 10 waves -> 10 tiles of 16 pixels = W
#define BN_EPS  1e-3f

typedef __attribute__((ext_vector_type(16))) _Float16 v16h;
typedef __attribute__((ext_vector_type(8)))  float    v8f;

union F16x2 { unsigned int u; _Float16 h[2]; };

__device__ inline unsigned int pack2(float a, float b) {
    F16x2 p; p.h[0] = (_Float16)a; p.h[1] = (_Float16)b; return p.u;
}

__global__ __launch_bounds__(BLOCK)
void stereo_attn_kernel(const float* __restrict__ inL,
                        const float* __restrict__ inR,
                        const float* __restrict__ kern,   // [K_TOTAL][MM] (HWIO)
                        const float* __restrict__ bias,   // [MM]
                        const float* __restrict__ gamma_,
                        const float* __restrict__ beta_,
                        const float* __restrict__ mean_,
                        const float* __restrict__ var_,
                        float* __restrict__ outX,
                        float* __restrict__ outW0,
                        float* __restrict__ outW1)
{
    // A-fragments (kernel^T, f16) laid out exactly as WMMA A registers:
    // chunk kc, lane L holds row m=L&15, element s holds k = 16*(L>>4) + s, contiguous order matching B.
    __shared__ v16h  sAfrag[KCH * 32];                 // 40 KB
    __shared__ __align__(16) float sW[WW][MM];         // softmax weights per pixel
    __shared__ __align__(16) float sScale[CC];
    __shared__ __align__(16) float sShift[CC];

    const int t   = threadIdx.x;
    const int row = blockIdx.x;            // n*H + h
    const size_t rowBase = (size_t)row * WW * CC;

    // ---- Stage 0a: BN affine precompute ----
    if (t < CC) {
        float sc = gamma_[t] * rsqrtf(var_[t] + BN_EPS);
        sScale[t] = sc;
        sShift[t] = beta_[t] - mean_[t] * sc;
    }

    // ---- Stage 0b: build A fragments in LDS (f16, transposed kernel) ----
    {
        unsigned int* aPack = (unsigned int*)sAfrag;
        for (int dw = t; dw < KCH * 32 * 8; dw += BLOCK) {
            int kc   = dw >> 8;        // 256 dwords per chunk (32 lanes x 8)
            int rem  = dw & 255;
            int lane = rem >> 3;
            int j    = rem & 7;
            int m    = lane & 15;
            int k0   = kc * 32 + ((lane & 16) ? 16 : 0) + 2 * j;
            float v0 = 0.f, v1 = 0.f;
            if (m < MM) { v0 = kern[k0 * MM + m]; v1 = kern[(k0 + 1) * MM + m]; }
            aPack[dw] = pack2(v0, v1);
        }
    }
    __syncthreads();

    // ---- Stage 1: conv logits via WMMA. D[m][pixel] over 16-pixel tile ----
    const int wave    = t >> 5;         // tile id 0..9
    const int lane    = t & 31;
    const int pInTile = lane & 15;
    const int hiHalf  = (lane >> 4) & 1;
    const int wp      = wave * 16 + pInTile;   // pixel 0..159

    v8f acc = {};
    for (int kc = 0; kc < KCH; ++kc) {
        v16h a = sAfrag[kc * 32 + lane];               // 2x ds_load_b128
        int kbase = kc * 32 + hiHalf * 16;             // this lane's 16 K values
        int tap   = kbase / TWO_C;
        int c0    = kbase % TWO_C;
        int wsrc  = wp + tap - (KW / 2);               // SAME zero padding
        v16h b;
        if (wsrc >= 0 && wsrc < WW) {
            const float* src = (c0 < CC)
                ? (inL + rowBase + (size_t)wsrc * CC + c0)
                : (inR + rowBase + (size_t)wsrc * CC + (c0 - CC));
            const float4* s4 = (const float4*)src;     // 16B aligned (c0 mult of 16)
            #pragma unroll
            for (int q = 0; q < 4; ++q) {
                float4 f = s4[q];
                b[4*q+0] = (_Float16)f.x; b[4*q+1] = (_Float16)f.y;
                b[4*q+2] = (_Float16)f.z; b[4*q+3] = (_Float16)f.w;
            }
        } else {
            #pragma unroll
            for (int q = 0; q < 16; ++q) b[q] = (_Float16)0.f;
        }
        acc = __builtin_amdgcn_wmma_f32_16x16x32_f16(
                  false, a, false, b, (short)0, acc, false, false);
    }

    // ---- Stage 2: bias + softmax over m (lane pair shares one pixel) ----
    // lo lane (hiHalf==0): m = 0..7 in acc[0..7]; hi lane: m = 8..9 in acc[0..1]
    float v[8];
    #pragma unroll
    for (int r = 0; r < 8; ++r) v[r] = acc[r];
    if (!hiHalf) {
        #pragma unroll
        for (int r = 0; r < 8; ++r) v[r] += bias[r];
    } else {
        v[0] += bias[8]; v[1] += bias[9];
    }
    const int nMine = hiHalf ? 2 : 8;
    float mx = -3.402823466e38f;
    for (int r = 0; r < nMine; ++r) mx = fmaxf(mx, v[r]);
    mx = fmaxf(mx, __shfl_xor(mx, 16, 32));
    float e[8];
    #pragma unroll
    for (int r = 0; r < 8; ++r) e[r] = 0.f;
    float sum = 0.f;
    for (int r = 0; r < nMine; ++r) { e[r] = __expf(v[r] - mx); sum += e[r]; }
    sum += __shfl_xor(sum, 16, 32);
    const float inv = 1.0f / sum;

    const size_t wBase = ((size_t)row * WW + wp) * MM;
    if (!hiHalf) {
        #pragma unroll
        for (int q = 0; q < 4; ++q) {
            float2 p; p.x = e[2*q] * inv; p.y = e[2*q+1] * inv;
            *(float2*)(outW0 + wBase + 2*q) = p;
            *(float2*)(outW1 + wBase + 2*q) = p;
            *(float2*)&sW[wp][2*q] = p;
        }
    } else {
        float2 p; p.x = e[0] * inv; p.y = e[1] * inv;
        *(float2*)(outW0 + wBase + 8) = p;
        *(float2*)(outW1 + wBase + 8) = p;
        *(float2*)&sW[wp][8] = p;
    }
    __syncthreads();

    // ---- Stage 3: x = sum_m w_m * shifted view (circular roll!), BN + tanh ----
    const int C4 = CC / 4;   // 32 float4 chunks per pixel
    for (int eIdx = t; eIdx < WW * C4; eIdx += BLOCK) {
        int w  = eIdx / C4;
        int c4 = (eIdx % C4) * 4;
        float4 av; av.x = 0.f; av.y = 0.f; av.z = 0.f; av.w = 0.f;
        #pragma unroll
        for (int i = 0; i < KW; ++i) {
            int wl = w - i; wl += (wl < 0)   ? WW : 0;   // roll(L, +i)
            int wr = w + i; wr -= (wr >= WW) ? WW : 0;   // roll(R, -i)
            float wlw = sW[w][2*i];
            float wrw = sW[w][2*i + 1];
            float4 fl = *(const float4*)(inL + rowBase + (size_t)wl * CC + c4);
            float4 fr = *(const float4*)(inR + rowBase + (size_t)wr * CC + c4);
            av.x = fmaf(wlw, fl.x, av.x); av.y = fmaf(wlw, fl.y, av.y);
            av.z = fmaf(wlw, fl.z, av.z); av.w = fmaf(wlw, fl.w, av.w);
            av.x = fmaf(wrw, fr.x, av.x); av.y = fmaf(wrw, fr.y, av.y);
            av.z = fmaf(wrw, fr.z, av.z); av.w = fmaf(wrw, fr.w, av.w);
        }
        float4 sc = *(const float4*)&sScale[c4];
        float4 sh = *(const float4*)&sShift[c4];
        float4 o;
        o.x = tanhf(fmaf(av.x, sc.x, sh.x));
        o.y = tanhf(fmaf(av.y, sc.y, sh.y));
        o.z = tanhf(fmaf(av.z, sc.z, sh.z));
        o.w = tanhf(fmaf(av.w, sc.w, sh.w));
        *(float4*)(outX + rowBase + (size_t)w * CC + c4) = o;
    }
}

extern "C" void kernel_launch(void* const* d_in, const int* in_sizes, int n_in,
                              void* d_out, int out_size, void* d_ws, size_t ws_size,
                              hipStream_t stream) {
    const float* inL    = (const float*)d_in[0];
    const float* inR    = (const float*)d_in[1];
    const float* kern   = (const float*)d_in[2];
    const float* bias   = (const float*)d_in[3];
    const float* gamma_ = (const float*)d_in[4];
    const float* beta_  = (const float*)d_in[5];
    const float* mean_  = (const float*)d_in[6];
    const float* var_   = (const float*)d_in[7];

    float* outX  = (float*)d_out;
    float* outW0 = outX  + (size_t)NB * HH * WW * CC;   // weights copy 1
    float* outW1 = outW0 + (size_t)NB * HH * WW * MM;   // weights copy 2

    dim3 grid(NB * HH);   // one block per (n, h) row
    dim3 block(BLOCK);
    stereo_attn_kernel<<<grid, block, 0, stream>>>(
        inL, inR, kern, bias, gamma_, beta_, mean_, var_, outX, outW0, outW1);
}
